// SwinUNet_7791070675430
// MI455X (gfx1250) — matrix operand order repaired and checked by reading, not compiled, not run on hardware
//
#include <hip/hip_runtime.h>
#include <hip/hip_bf16.h>
#include <math.h>

typedef __attribute__((ext_vector_type(16))) _Float16 v16h;
typedef __attribute__((ext_vector_type(8)))  float    v8f;
typedef int v4i __attribute__((vector_size(4 * sizeof(int))));

#define AS1 __attribute__((address_space(1)))
#define AS3 __attribute__((address_space(3)))

#if __has_builtin(__builtin_amdgcn_global_load_async_to_lds_b128) && \
    __has_builtin(__builtin_amdgcn_s_wait_asynccnt)
#define USE_ASYNC_LDS 1
#else
#define USE_ASYNC_LDS 0
#endif

__device__ __forceinline__ v8f wmma_f16(v16h a, v16h b, v8f c) {
    // D = A(16x32 f16) * B(32x16 f16) + C(16x16 f32)
    return __builtin_amdgcn_wmma_f32_16x16x32_f16(false, a, false, b, (short)0, c, false, false);
}

__device__ __forceinline__ void async_cp_b128(const float* g, float* l) {
#if USE_ASYNC_LDS
    __builtin_amdgcn_global_load_async_to_lds_b128((AS1 v4i*)g, (AS3 v4i*)l, 0, 0);
#else
    *(float4*)l = *(const float4*)g;
#endif
}

__device__ __forceinline__ void async_wait() {
#if USE_ASYNC_LDS
    __builtin_amdgcn_s_wait_asynccnt(0);
#endif
}

// ---------------------------------------------------------------------------
// Generic GEMM: C[M,N] = act(A[M,K] @ W[K,N] + bias) + residual
// 4 waves / block; wave w computes rows [tm0 + 16w, tm0 + 16w + 16) of one
// 16-wide N tile. A (64x32 f32) and W (32x16 f32) chunks staged in LDS via
// async-to-LDS (ASYNCcnt) when available. M % 64 == 0, N % 16, K % 32.
// act: 0 = none, 1 = exact GELU
// ---------------------------------------------------------------------------
__global__ void gemm_wmma_kernel(const float* __restrict__ A, const float* __restrict__ W,
                                 const float* __restrict__ bias, const float* __restrict__ res,
                                 float* __restrict__ C, int M, int N, int K, int act) {
    __shared__ float As[64 * 32];   // [row][k]   8 KB
    __shared__ float Bs[32 * 16];   // [k][col]   2 KB
    const int tid  = threadIdx.x;
    const int lane = tid & 31;
    const int w    = tid >> 5;
    const int tn   = blockIdx.x * 16;
    const int tm0  = blockIdx.y * 64;
    const int tm   = tm0 + w * 16;
    const int l15  = lane & 15;
    const int lhi  = (lane & 16) ? 1 : 0;
    const int kAoff = lhi ? 8 : 0;
    const int kBoff = lhi ? 16 : 0;

    v8f acc = {};
    for (int k0 = 0; k0 < K; k0 += 32) {
        __syncthreads();   // previous iteration's LDS reads complete
#pragma unroll
        for (int j = 0; j < 4; ++j) {         // A chunk: 512 float4
            int q = j * 128 + tid;
            int row = q >> 3, f4 = q & 7;
            async_cp_b128(A + (size_t)(tm0 + row) * K + k0 + f4 * 4,
                          As + row * 32 + f4 * 4);
        }
        {                                     // W chunk: 128 float4
            int row = tid >> 2, f4 = tid & 3;
            async_cp_b128(W + (size_t)(k0 + row) * N + tn + f4 * 4,
                          Bs + row * 16 + f4 * 4);
        }
        async_wait();
        __syncthreads();   // staging visible to all waves

        v16h a, b;
#pragma unroll
        for (int t = 0; t < 16; ++t) {
            int kk = t + kBoff;     // A frag layout (16-bit 16x32)
            a[t] = (_Float16)As[(w * 16 + l15) * 32 + kk];
        }
#pragma unroll
        for (int t = 0; t < 16; ++t) {
            b[t] = (_Float16)Bs[(t + kBoff) * 16 + l15];    // B frag layout (16-bit 32x16)
        }
        acc = wmma_f16(a, b, acc);
    }

    const int col = tn + l15;
    const int rb  = tm + (lhi ? 8 : 0);
#pragma unroll
    for (int r = 0; r < 8; ++r) {
        float v = acc[r];
        if (bias) v += bias[col];
        if (act == 1) v = 0.5f * v * (1.0f + erff(v * 0.70710678118654752f));
        if (res) v += res[(size_t)(rb + r) * N + col];
        C[(size_t)(rb + r) * N + col] = v;
    }
}

// ---------------------------------------------------------------------------
// Flash attention with inline relative-position bias. head_dim == 32 always.
// qkv: [N, 3C] (q at h*32, k at C + h*32, v at 2C + h*32). out: [N, C].
// grid = (N/64, nheads), block = 128: 4 waves share K/V chunk staging in LDS,
// each wave owns one 16-query tile. Never materializes the NxN score matrix.
// ---------------------------------------------------------------------------
__global__ void attn_flash_kernel(const float* __restrict__ qkv, const float* __restrict__ rpb,
                                  float* __restrict__ outp, int N, int Cc, int nh, int wsz) {
    __shared__ float    Ks[32 * 32];        // [key][d] 4 KB
    __shared__ float    Vs[32 * 32];        // [key][d] 4 KB
    __shared__ _Float16 P[4][16 * 32];      // per-wave P relayout buffer, 4 KB
    const int tid  = threadIdx.x;
    const int lane = tid & 31;
    const int w    = tid >> 5;
    const int h    = blockIdx.y;
    const int q0   = blockIdx.x * 64 + w * 16;
    const int C3   = 3 * Cc;
    const int hb   = h * 32;
    const int l15  = lane & 15;
    const int lhi  = (lane & 16) ? 1 : 0;
    const int kAoff = lhi ? 8 : 0;
    const int rbase = lhi ? 8 : 0;
    const int dOff  = lhi ? 16 : 0;
    const int tw    = 2 * wsz - 1;
    const float scale = 0.17677669529663688f; // 32^-0.5

    v16h qf;
    {
        const float* qp = qkv + (size_t)(q0 + l15) * C3 + hb;
#pragma unroll
        for (int t = 0; t < 16; ++t) {
            int kk = t + dOff;
            qf[t] = (_Float16)(qp[kk] * scale);
        }
    }

    v8f olo = {}, ohi = {};
    float mrow[8], lrow[8];
#pragma unroll
    for (int r = 0; r < 8; ++r) { mrow[r] = -1e30f; lrow[r] = 0.0f; }

    for (int kc = 0; kc < N; kc += 32) {
        __syncthreads();   // previous iteration's LDS reads complete
#pragma unroll
        for (int j = 0; j < 2; ++j) {         // K and V chunks: 256 float4 each
            int q = j * 128 + tid;
            int row = q >> 3, f4 = q & 7;
            async_cp_b128(qkv + (size_t)(kc + row) * C3 + Cc + hb + f4 * 4,
                          Ks + row * 32 + f4 * 4);
            async_cp_b128(qkv + (size_t)(kc + row) * C3 + 2 * Cc + hb + f4 * 4,
                          Vs + row * 32 + f4 * 4);
        }
        async_wait();
        __syncthreads();   // staging visible

        // ---- S = Q @ K^T for 32 keys (two 16x16 tiles) ----
        v16h kf0, kf1;
#pragma unroll
        for (int t = 0; t < 16; ++t) {
            kf0[t] = (_Float16)Ks[l15 * 32 + t + dOff];
            kf1[t] = (_Float16)Ks[(l15 + 16) * 32 + t + dOff];
        }
        v8f z = {};
        v8f s0 = wmma_f16(qf, kf0, z);
        v8f s1 = wmma_f16(qf, kf1, z);

        // ---- bias + online softmax (D-layout: col = l15, rows rbase..rbase+7) ----
        const int col0 = kc + l15, col1 = col0 + 16;
        const int ky0 = col0 / wsz, kx0 = col0 % wsz;
        const int ky1 = col1 / wsz, kx1 = col1 % wsz;
        float p0[8], p1[8];
#pragma unroll
        for (int r = 0; r < 8; ++r) {
            const int qr = q0 + rbase + r;
            const int qy = qr / wsz, qx = qr % wsz;
            const int i0 = (qy - ky0 + wsz - 1) * tw + (qx - kx0 + wsz - 1);
            const int i1 = (qy - ky1 + wsz - 1) * tw + (qx - kx1 + wsz - 1);
            float a = s0[r] + rpb[(size_t)i0 * nh + h];
            float b = s1[r] + rpb[(size_t)i1 * nh + h];
            float mx = fmaxf(a, b);
            for (int m = 1; m < 16; m <<= 1) mx = fmaxf(mx, __shfl_xor(mx, m, 32));
            float mnew = fmaxf(mrow[r], mx);
            float al = __expf(mrow[r] - mnew);
            float e0 = __expf(a - mnew), e1 = __expf(b - mnew);
            float sum = e0 + e1;
            for (int m = 1; m < 16; m <<= 1) sum += __shfl_xor(sum, m, 32);
            lrow[r] = lrow[r] * al + sum;
            mrow[r] = mnew;
            olo[r] *= al;
            ohi[r] *= al;
            p0[r] = e0; p1[r] = e1;
        }

        // ---- relayout P (D-layout -> A-layout) through per-wave LDS ----
#pragma unroll
        for (int r = 0; r < 8; ++r) {
            P[w][(rbase + r) * 32 + l15]      = (_Float16)p0[r];
            P[w][(rbase + r) * 32 + l15 + 16] = (_Float16)p1[r];
        }
        __syncthreads();   // P visible (uniform barrier across all 4 waves)
        v16h pf;
#pragma unroll
        for (int t = 0; t < 16; ++t) {
            int kk = t + dOff;
            pf[t] = P[w][l15 * 32 + kk];
        }

        // ---- O += P @ V (two 16x16 output halves over d) ----
        v16h vf0, vf1;
#pragma unroll
        for (int t = 0; t < 16; ++t) {
            vf0[t] = (_Float16)Vs[(t + dOff) * 32 + l15];
            vf1[t] = (_Float16)Vs[(t + dOff) * 32 + 16 + l15];
        }
        olo = wmma_f16(pf, vf0, olo);
        ohi = wmma_f16(pf, vf1, ohi);
    }

#pragma unroll
    for (int r = 0; r < 8; ++r) {
        float inv = 1.0f / lrow[r];
        float* orow = outp + (size_t)(q0 + rbase + r) * Cc + hb;
        orow[l15]      = olo[r] * inv;
        orow[16 + l15] = ohi[r] * inv;
    }
}

// ---------------------------------------------------------------------------
// LayerNorm over last dim; one wave per token.
// ---------------------------------------------------------------------------
__global__ void ln_kernel(const float* __restrict__ x, const float* __restrict__ g,
                          const float* __restrict__ b, float* __restrict__ y, int C) {
    const int tok = blockIdx.x, lane = threadIdx.x;
    const float* row = x + (size_t)tok * C;
    float s = 0.f, ss = 0.f;
    for (int c = lane; c < C; c += 32) { float v = row[c]; s += v; ss += v * v; }
    for (int m = 16; m; m >>= 1) { s += __shfl_xor(s, m, 32); ss += __shfl_xor(ss, m, 32); }
    float mean = s / C;
    float var  = ss / C - mean * mean;
    float r    = rsqrtf(var + 1e-5f);
    float* out = y + (size_t)tok * C;
    for (int c = lane; c < C; c += 32) out[c] = (row[c] - mean) * r * g[c] + b[c];
}

// 4x4 stride-4 patch embed: x(3,256,256) -> t(4096, 96) token-major
__global__ void patch_embed_kernel(const float* __restrict__ x, const float* __restrict__ w,
                                   const float* __restrict__ b, float* __restrict__ t) {
    int i = blockIdx.x * blockDim.x + threadIdx.x;
    if (i >= 4096 * 96) return;
    int o = i % 96, tok = i / 96;
    int ty = tok >> 6, tx = tok & 63;
    float acc = b[o];
    for (int ic = 0; ic < 3; ++ic)
        for (int kh = 0; kh < 4; ++kh)
            for (int kw = 0; kw < 4; ++kw)
                acc += x[ic * 65536 + (ty * 4 + kh) * 256 + (tx * 4 + kw)] *
                       w[((o * 3 + ic) * 4 + kh) * 4 + kw];
    t[(size_t)tok * 96 + o] = acc;
}

// patch-merge gather: t(H*W, C) -> out(H/2*W/2, 4C), order (0,0),(1,0),(0,1),(1,1)
__global__ void merge_cat_kernel(const float* __restrict__ t, float* __restrict__ outp,
                                 int H, int W, int Cc) {
    int total = (H / 2) * (W / 2) * 4 * Cc;
    int i = blockIdx.x * blockDim.x + threadIdx.x;
    if (i >= total) return;
    int c4 = i % (4 * Cc), tok = i / (4 * Cc);
    int ox = tok % (W / 2), oy = tok / (W / 2);
    int part = c4 / Cc, c = c4 % Cc;
    int dy = (part == 1 || part == 3) ? 1 : 0;
    int dx = (part >= 2) ? 1 : 0;
    outp[i] = t[((size_t)(2 * oy + dy) * W + (2 * ox + dx)) * Cc + c];
}

// 2x bilinear upsample (align_corners=True) on channel-last tokens + skip add
__global__ void up_bilinear_add_kernel(const float* __restrict__ tin, const float* __restrict__ skip,
                                       float* __restrict__ outp, int H, int W, int Cc) {
    const int OH = 2 * H, OW = 2 * W;
    int total = OH * OW * Cc;
    int i = blockIdx.x * blockDim.x + threadIdx.x;
    if (i >= total) return;
    int c = i % Cc, ox = (i / Cc) % OW, oy = i / (Cc * OW);
    float sy = oy * (float)(H - 1) / (float)(OH - 1);
    float sx = ox * (float)(W - 1) / (float)(OW - 1);
    int y0 = (int)floorf(sy); int y1 = min(y0 + 1, H - 1); float fy = sy - y0;
    int x0 = (int)floorf(sx); int x1 = min(x0 + 1, W - 1); float fx = sx - x0;
    float v00 = tin[((size_t)y0 * W + x0) * Cc + c];
    float v01 = tin[((size_t)y0 * W + x1) * Cc + c];
    float v10 = tin[((size_t)y1 * W + x0) * Cc + c];
    float v11 = tin[((size_t)y1 * W + x1) * Cc + c];
    float top = v00 * (1.f - fx) + v01 * fx;
    float bot = v10 * (1.f - fx) + v11 * fx;
    outp[i] = top * (1.f - fy) + bot * fy + skip[i];
}

// ConvTranspose2d 2x2 stride 2: t(4096, 96) -> img(128*128, 48) channel-last
__global__ void convt2x2_kernel(const float* __restrict__ t, const float* __restrict__ w,
                                const float* __restrict__ b, float* __restrict__ outp) {
    int i = blockIdx.x * blockDim.x + threadIdx.x;
    if (i >= 128 * 128 * 48) return;
    int o = i % 48, ox = (i / 48) % 128, oy = i / (48 * 128);
    int y = oy >> 1, dh = oy & 1, x = ox >> 1, dw = ox & 1;
    const float* trow = t + ((size_t)y * 64 + x) * 96;
    float acc = b[o];
    for (int c = 0; c < 96; ++c)
        acc += trow[c] * w[((c * 48 + o) * 2 + dh) * 2 + dw];
    outp[i] = acc;
}

// bilinear 128->256 (align_corners) fused with final 1x1 conv (48 -> 1)
__global__ void final_head_kernel(const float* __restrict__ img, const float* __restrict__ fcw,
                                  const float* __restrict__ fcb, float* __restrict__ outp) {
    int i = blockIdx.x * blockDim.x + threadIdx.x;
    if (i >= 256 * 256) return;
    int ox = i % 256, oy = i / 256;
    float sy = oy * 127.0f / 255.0f;
    float sx = ox * 127.0f / 255.0f;
    int y0 = (int)floorf(sy); int y1 = min(y0 + 1, 127); float fy = sy - y0;
    int x0 = (int)floorf(sx); int x1 = min(x0 + 1, 127); float fx = sx - x0;
    float w00 = (1.f - fy) * (1.f - fx), w01 = (1.f - fy) * fx;
    float w10 = fy * (1.f - fx),         w11 = fy * fx;
    const float* p00 = img + ((size_t)y0 * 128 + x0) * 48;
    const float* p01 = img + ((size_t)y0 * 128 + x1) * 48;
    const float* p10 = img + ((size_t)y1 * 128 + x0) * 48;
    const float* p11 = img + ((size_t)y1 * 128 + x1) * 48;
    float acc = fcb[0];
    for (int c = 0; c < 48; ++c)
        acc += fcw[c] * (w00 * p00[c] + w01 * p01[c] + w10 * p10[c] + w11 * p11[c]);
    outp[i] = acc;
}

// ---------------------------------------------------------------------------
struct BP {
    const float *ln1_g, *ln1_b, *qkv_w, *rpb, *proj_w, *proj_b;
    const float *ln2_g, *ln2_b, *fc1_w, *fc1_b, *fc2_w, *fc2_b;
};

extern "C" void kernel_launch(void* const* d_in, const int* in_sizes, int n_in,
                              void* d_out, int out_size, void* d_ws, size_t ws_size,
                              hipStream_t stream) {
    (void)in_sizes; (void)n_in; (void)out_size; (void)ws_size;
    int idx = 0;
    auto nxt = [&]() { return (const float*)d_in[idx++]; };

    const float* x    = nxt();
    const float* pe_w = nxt();
    const float* pe_b = nxt();

    BP sblk[4][2];
    const float *mg_g[4], *mg_b[4], *mg_w[4];
    auto readBP = [&](BP& p) {
        p.ln1_g = nxt(); p.ln1_b = nxt(); p.qkv_w = nxt(); p.rpb = nxt();
        p.proj_w = nxt(); p.proj_b = nxt(); p.ln2_g = nxt(); p.ln2_b = nxt();
        p.fc1_w = nxt(); p.fc1_b = nxt(); p.fc2_w = nxt(); p.fc2_b = nxt();
    };
    for (int s = 0; s < 4; ++s) {
        readBP(sblk[s][0]); readBP(sblk[s][1]);
        mg_g[s] = nxt(); mg_b[s] = nxt(); mg_w[s] = nxt();
    }
    const float* up4_w = nxt(); BP ub4; readBP(ub4);
    const float* up3_w = nxt(); BP ub3; readBP(ub3);
    const float* up2_w = nxt(); BP ub2; readBP(ub2);
    const float* up1_w = nxt(); const float* up1_b = nxt();
    const float* fc_w  = nxt(); const float* fc_b  = nxt();

    // workspace arena (floats)
    float* wsf = (float*)d_ws;
    size_t off = 0;
    auto wsalloc = [&](size_t n) { float* p = wsf + off; off += n; return p; };
    float* T   = wsalloc(4096 * 96);
    float* TMP = wsalloc(4096 * 96);
    float* T2  = wsalloc(4096 * 96);
    float* QKV = wsalloc(4096 * 288);
    float* ATT = wsalloc(4096 * 96);
    float* HID = wsalloc(4096 * 384);
    float* S[4];
    S[0] = wsalloc(4096 * 96);  S[1] = wsalloc(1024 * 192);
    S[2] = wsalloc(256 * 384);  S[3] = wsalloc(64 * 768);
    float* IMG = wsalloc(128 * 128 * 48);

    auto gemm = [&](const float* A, const float* Wm, const float* bias, const float* res,
                    float* Cout, int M, int Nn, int K, int act) {
        dim3 g(Nn / 16, M / 64);
        gemm_wmma_kernel<<<g, 128, 0, stream>>>(A, Wm, bias, res, Cout, M, Nn, K, act);
    };
    auto elems = [&](int total) { return dim3((total + 255) / 256); };

    auto run_block = [&](float* t, int Ntok, int Cc, int nh, int wsz, const BP& p) {
        ln_kernel<<<Ntok, 32, 0, stream>>>(t, p.ln1_g, p.ln1_b, TMP, Cc);
        gemm(TMP, p.qkv_w, nullptr, nullptr, QKV, Ntok, 3 * Cc, Cc, 0);
        attn_flash_kernel<<<dim3(Ntok / 64, nh), 128, 0, stream>>>(QKV, p.rpb, ATT, Ntok, Cc, nh, wsz);
        gemm(ATT, p.proj_w, p.proj_b, t, t, Ntok, Cc, Cc, 0);
        ln_kernel<<<Ntok, 32, 0, stream>>>(t, p.ln2_g, p.ln2_b, TMP, Cc);
        gemm(TMP, p.fc1_w, p.fc1_b, nullptr, HID, Ntok, 4 * Cc, Cc, 1);
        gemm(HID, p.fc2_w, p.fc2_b, t, t, Ntok, Cc, 4 * Cc, 0);
    };

    // ---- encoder ----
    patch_embed_kernel<<<elems(4096 * 96), 256, 0, stream>>>(x, pe_w, pe_b, T);
    const int dims[4]  = {96, 192, 384, 768};
    const int heads[4] = {3, 6, 12, 24};
    const int wss[4]   = {64, 32, 16, 8};
    int H = 64, W = 64;
    for (int s = 0; s < 4; ++s) {
        const int Ntok = H * W, Cc = dims[s];
        run_block(T, Ntok, Cc, heads[s], wss[s], sblk[s][0]);
        run_block(T, Ntok, Cc, heads[s], wss[s], sblk[s][1]);
        (void)hipMemcpyAsync(S[s], T, (size_t)Ntok * Cc * sizeof(float),
                             hipMemcpyDeviceToDevice, stream);
        if (s < 3) {
            merge_cat_kernel<<<elems(Ntok * Cc), 256, 0, stream>>>(T, T2, H, W, Cc);
            ln_kernel<<<Ntok / 4, 32, 0, stream>>>(T2, mg_g[s], mg_b[s], TMP, 4 * Cc);
            gemm(TMP, mg_w[s], nullptr, nullptr, T, Ntok / 4, 2 * Cc, 4 * Cc, 0);
            H /= 2; W /= 2;
        }
    }

    // ---- decoder ----
    gemm(S[3], up4_w, nullptr, nullptr, T2, 64, 384, 768, 0);
    up_bilinear_add_kernel<<<elems(256 * 384), 256, 0, stream>>>(T2, S[2], T, 8, 8, 384);
    run_block(T, 256, 384, 12, 16, ub4);

    gemm(T, up3_w, nullptr, nullptr, T2, 256, 192, 384, 0);
    up_bilinear_add_kernel<<<elems(1024 * 192), 256, 0, stream>>>(T2, S[1], T, 16, 16, 192);
    run_block(T, 1024, 192, 6, 32, ub3);

    gemm(T, up2_w, nullptr, nullptr, T2, 1024, 96, 192, 0);
    up_bilinear_add_kernel<<<elems(4096 * 96), 256, 0, stream>>>(T2, S[0], T, 32, 32, 96);
    run_block(T, 4096, 96, 3, 64, ub2);

    // ---- head ----
    convt2x2_kernel<<<elems(128 * 128 * 48), 256, 0, stream>>>(T, up1_w, up1_b, IMG);
    final_head_kernel<<<elems(256 * 256), 256, 0, stream>>>(IMG, fc_w, fc_b, (float*)d_out);
}
